// GCNGuard_88313117541059
// MI455X (gfx1250) — matrix operand order repaired and checked by reading, not compiled, not run on hardware
//
#include <hip/hip_runtime.h>
#include <math.h>
#include <stdint.h>

// ---------------------------------------------------------------------------
// CDNA5 / gfx1250 implementation of GCNGuard forward.
// wave32 everywhere; GEMMs via v_wmma_f32_16x16x32_bf16 with pre-swizzled B
// and LDS-transposed b128 stores.
// ---------------------------------------------------------------------------

typedef __attribute__((ext_vector_type(16))) __bf16 v16bf;
typedef __attribute__((ext_vector_type(8)))  float  v8f;

#define WF 32
#define COS_EPS 1e-8f
#define THRESH  0.1f
#define CSTRIDE 20            // LDS row stride (floats): conflict-free, 16B aligned

__device__ __forceinline__ unsigned short f2bf(float f) {
  // round-to-nearest-even f32 -> bf16
  unsigned int u = __builtin_bit_cast(unsigned int, f);
  u += 0x7FFFu + ((u >> 16) & 1u);
  return (unsigned short)(u >> 16);
}

__device__ __forceinline__ float wave_sum(float v) {
#pragma unroll
  for (int off = 16; off > 0; off >>= 1) v += __shfl_xor(v, off, WF);
  return v;
}
__device__ __forceinline__ float wave_max(float v) {
#pragma unroll
  for (int off = 16; off > 0; off >>= 1) v = fmaxf(v, __shfl_xor(v, off, WF));
  return v;
}

// ---------------------------------------------------------------------------
__global__ void zero_f32_kernel(float* __restrict__ p, int n) {
  int i = blockIdx.x * blockDim.x + threadIdx.x;
  if (i < n) p[i] = 0.0f;
}

// one wave per row: nrm[r] = ||X[r,:]||_2
__global__ void row_norm_kernel(const float* __restrict__ X, float* __restrict__ nrm,
                                int Nn, int D) {
  int wid  = (blockIdx.x * blockDim.x + threadIdx.x) >> 5;
  int lane = threadIdx.x & 31;
  if (wid >= Nn) return;
  const float* xr = X + (size_t)wid * D;
  float s = 0.0f;
  for (int d = lane * 4; d < D; d += 128) {
    float4 v = *(const float4*)(xr + d);
    s += v.x * v.x + v.y * v.y + v.z * v.z + v.w * v.w;
  }
  s = wave_sum(s);
  if (lane == 0) nrm[wid] = sqrtf(s);
}

// one wave per edge: thresholded cosine similarity + segment sums over col.
// edges e >= Eorig are implicit self loops (r = c = e - Eorig).
__global__ void edge_cos_kernel(const float* __restrict__ X, const float* __restrict__ nrm,
                                const int* __restrict__ rowI, const int* __restrict__ colI,
                                float* __restrict__ s, float* __restrict__ ssum,
                                float* __restrict__ degc, int nE, int Eorig, int D) {
  int wid  = (blockIdx.x * blockDim.x + threadIdx.x) >> 5;
  int lane = threadIdx.x & 31;
  if (wid >= nE) return;
  int r, c;
  if (wid < Eorig) { r = rowI[wid]; c = colI[wid]; }
  else             { r = c = wid - Eorig; }
  const float* xr = X + (size_t)r * D;
  const float* xc = X + (size_t)c * D;
  float dot = 0.0f;
  for (int d = lane * 4; d < D; d += 128) {
    float4 a = *(const float4*)(xr + d);
    float4 b = *(const float4*)(xc + d);
    dot += a.x * b.x + a.y * b.y + a.z * b.z + a.w * b.w;
  }
  dot = wave_sum(dot);
  if (lane == 0) {
    float nu = fmaxf(nrm[r], COS_EPS);
    float nv = fmaxf(nrm[c], COS_EPS);
    float sv = dot / (nu * nv);
    if (sv < THRESH) sv = 1e-5f;
    s[wid] = sv;
    atomicAdd(&ssum[c], sv);
    if (degc) atomicAdd(&degc[c], 1.0f);
  }
}

// layer-0 weights: normalize by ssum[row], self-loop rescale, exp()
__global__ void edge_weight0_kernel(const float* __restrict__ s, const float* __restrict__ ssum,
                                    const float* __restrict__ degc, const int* __restrict__ rowI,
                                    float* __restrict__ ew, int E, int Nn) {
  int e = blockIdx.x * blockDim.x + threadIdx.x;
  if (e >= E + Nn) return;
  float sv;
  if (e < E) {
    int r = rowI[e];
    float d = degc[r];
    sv = (s[e] / ssum[r]) * (d / (d + 1.0f));
  } else {
    int v = e - E;
    sv = 1.0f / (degc[v] + 1.0f);
  }
  ew[e] = expf(sv);
}

// layer-1 weights + gated blend with memory
__global__ void edge_weight1_kernel(const float* __restrict__ s2, const float* __restrict__ ssum2,
                                    const int* __restrict__ rowI, const float* __restrict__ ew_mem,
                                    const float* __restrict__ gate, float* __restrict__ ewb,
                                    int E, int Nn) {
  int e = blockIdx.x * blockDim.x + threadIdx.x;
  if (e >= E + Nn) return;
  int r = (e < E) ? rowI[e] : (e - E);
  float ew1 = expf(s2[e] / ssum2[r]);
  float g = gate[0];
  ewb[e] = g * ew_mem[e] + (1.0f - g) * ew1;
}

// degw[c] += ew[e]  (weighted in-degree for gcn_norm)
__global__ void deg_accum_kernel(const float* __restrict__ ew, const int* __restrict__ colI,
                                 float* __restrict__ degw, int nE, int Eorig) {
  int e = blockIdx.x * blockDim.x + threadIdx.x;
  if (e >= nE) return;
  int c = (e < Eorig) ? colI[e] : (e - Eorig);
  atomicAdd(&degw[c], ew[e]);
}

__global__ void dis_kernel(const float* __restrict__ degw, float* __restrict__ dis, int Nn) {
  int v = blockIdx.x * blockDim.x + threadIdx.x;
  if (v >= Nn) return;
  float d = degw[v];
  dis[v] = (d > 0.0f) ? rsqrtf(d) : 0.0f;
}

// ---------------------------------------------------------------------------
// Pre-swizzle B[K,Ncols] (f32) into WMMA B-fragment order as packed bf16
// pairs: element (tn, ks, lane, r) holds B values for
//   n = tn*16 + (lane&15), k = ks*32 + (lane>>4)*8 + (r>>2)*8 + 2r  (k+1 in hi half)
// so each GEMM lane reads its 8-dword fragment with two contiguous uint4 loads.
// ---------------------------------------------------------------------------
__global__ void pack_B_kernel(const float* __restrict__ B, unsigned int* __restrict__ Bpack,
                              int K, int Ncols, int tilesN) {
  int idx = blockIdx.x * blockDim.x + threadIdx.x;
  int nk = K >> 5;
  int total = tilesN * nk * 32 * 8;
  if (idx >= total) return;
  int r    = idx & 7;
  int lane = (idx >> 3) & 31;
  int rest = idx >> 8;
  int ks   = rest % nk;
  int tn   = rest / nk;
  int n = tn * 16 + (lane & 15);
  int k = ks * 32 + (lane >> 4) * 8 + (r >> 2) * 8 + 2 * r;
  float f0 = 0.0f, f1 = 0.0f;
  if (n < Ncols) {
    f0 = B[(size_t)k * Ncols + n];
    f1 = B[(size_t)(k + 1) * Ncols + n];
  }
  Bpack[idx] = (unsigned int)f2bf(f0) | ((unsigned int)f2bf(f1) << 16);
}

// ---------------------------------------------------------------------------
// GEMM C[M,Ncols] = A[M,K] @ B, f32 in/out, bf16 WMMA compute.
// One wave per 16x16 C tile; B pre-swizzled; result staged through LDS so
// each lane stores one half-row of C with two b128 stores.
// KSTEPS>0: compile-time K/32 (fully unrolled); KSTEPS==0: runtime loop.
// ---------------------------------------------------------------------------
template <int KSTEPS>
__global__ void gemm_bf16_wmma_kernel(const float* __restrict__ A,
                                      const unsigned int* __restrict__ Bpack,
                                      float* __restrict__ C, int M, int K, int Ncols,
                                      int tilesN, int numTiles) {
  __shared__ float ldsC[8 * 16 * CSTRIDE];          // 8 waves/block, 16x16 tile each
  int wid  = (blockIdx.x * blockDim.x + threadIdx.x) >> 5;
  int lane = threadIdx.x & 31;
  if (wid >= numTiles) return;
  int tm = wid / tilesN, tn = wid - tm * tilesN;
  int m0 = tm * 16, n0 = tn * 16;
  int half = lane >> 4;
  int mr   = lane & 15;

  int arowi = m0 + mr; if (arowi >= M) arowi = M - 1;   // clamp (store guarded)
  const float* arow = A + (size_t)arowi * K + 8 * half;

  const int nk = (KSTEPS > 0) ? KSTEPS : (K >> 5);
  const uint4* bp = (const uint4*)(Bpack) + ((tn * nk) * 32 + lane) * 2;

  v8f acc = {};
  auto kstep = [&](int ks) {
    union { v16bf v; unsigned short s[16]; unsigned int u[8]; } a, b;
    // A fragment: 4 x float4 contiguous loads
    const float* ap = arow + ks * 32;
    float4 a0 = *(const float4*)(ap);
    float4 a1 = *(const float4*)(ap + 4);
    float4 a2 = *(const float4*)(ap + 16);
    float4 a3 = *(const float4*)(ap + 20);
    a.s[0]  = f2bf(a0.x); a.s[1]  = f2bf(a0.y); a.s[2]  = f2bf(a0.z); a.s[3]  = f2bf(a0.w);
    a.s[4]  = f2bf(a1.x); a.s[5]  = f2bf(a1.y); a.s[6]  = f2bf(a1.z); a.s[7]  = f2bf(a1.w);
    a.s[8]  = f2bf(a2.x); a.s[9]  = f2bf(a2.y); a.s[10] = f2bf(a2.z); a.s[11] = f2bf(a2.w);
    a.s[12] = f2bf(a3.x); a.s[13] = f2bf(a3.y); a.s[14] = f2bf(a3.z); a.s[15] = f2bf(a3.w);
    // B fragment: 2 x uint4 contiguous loads from the pre-swizzled buffer
    const uint4* bq = bp + ks * 64;               // 32 lanes * 2 uint4 per k-step
    uint4 p0 = bq[0];
    uint4 p1 = bq[1];
    b.u[0] = p0.x; b.u[1] = p0.y; b.u[2] = p0.z; b.u[3] = p0.w;
    b.u[4] = p1.x; b.u[5] = p1.y; b.u[6] = p1.z; b.u[7] = p1.w;
    acc = __builtin_amdgcn_wmma_f32_16x16x32_bf16(false, a.v, false, b.v,
                                                  (short)0, acc, false, false);
  };
  if constexpr (KSTEPS > 0) {
#pragma unroll
    for (int ks = 0; ks < KSTEPS; ++ks) kstep(ks);
  } else {
    for (int ks = 0; ks < nk; ++ks) kstep(ks);
  }

  // ---- stage tile in LDS, then contiguous b128 row stores ----
  float* tl = &ldsC[(threadIdx.x >> 5) * 16 * CSTRIDE];
#pragma unroll
  for (int r = 0; r < 8; ++r)
    tl[(r + 8 * half) * CSTRIDE + mr] = acc[r];     // D layout: M=r+8h, N=lane&15
  __asm__ volatile("" ::: "memory");                // same-wave DS ops are in-order

  int rrow = lane >> 1;                             // 0..15
  int rcol = (lane & 1) * 8;                        // 0 or 8
  float4 c0 = *(const float4*)(tl + rrow * CSTRIDE + rcol);
  float4 c1 = *(const float4*)(tl + rrow * CSTRIDE + rcol + 4);
  int mrow = m0 + rrow;
  int cc   = n0 + rcol;
  if (mrow < M) {
    float* cp = C + (unsigned)(mrow * Ncols + cc);  // 32-bit index math
    if (cc + 7 < Ncols) {
      *(float4*)cp       = c0;
      *(float4*)(cp + 4) = c1;
    } else {                                        // generic partial-tile tail
      float vals[8] = {c0.x, c0.y, c0.z, c0.w, c1.x, c1.y, c1.z, c1.w};
      for (int j = 0; j < 8; ++j)
        if (cc + j < Ncols) cp[j] = vals[j];
    }
  }
}

// agg[c,:] += dis[r]*ew[e]*dis[c] * h[r,:]  — one wave per edge
__global__ void aggregate_kernel(const float* __restrict__ h, const float* __restrict__ ew,
                                 const float* __restrict__ dis, const int* __restrict__ rowI,
                                 const int* __restrict__ colI, float* __restrict__ agg,
                                 int nE, int Eorig, int D) {
  int wid  = (blockIdx.x * blockDim.x + threadIdx.x) >> 5;
  int lane = threadIdx.x & 31;
  if (wid >= nE) return;
  int r, c;
  if (wid < Eorig) { r = rowI[wid]; c = colI[wid]; }
  else             { r = c = wid - Eorig; }
  float w = dis[r] * ew[wid] * dis[c];
  const float* hr = h + (size_t)r * D;
  float* ac = agg + (size_t)c * D;
  for (int d = lane * 4; d < D; d += 128) {
    float4 hv = *(const float4*)(hr + d);
    atomicAdd(ac + d + 0, w * hv.x);
    atomicAdd(ac + d + 1, w * hv.y);
    atomicAdd(ac + d + 2, w * hv.z);
    atomicAdd(ac + d + 3, w * hv.w);
  }
}

__global__ void finalize_hidden_kernel(const float* __restrict__ agg, const float* __restrict__ b,
                                       float* __restrict__ x1, int total, int D) {
  int i = blockIdx.x * blockDim.x + threadIdx.x;
  if (i >= total) return;
  x1[i] = fmaxf(agg[i] + b[i % D], 0.0f);
}

// one wave per row: out = log_softmax(agg + b1), D <= 64
__global__ void logsoftmax_kernel(const float* __restrict__ agg, const float* __restrict__ b,
                                  float* __restrict__ out, int Nn, int D) {
  int wid  = (blockIdx.x * blockDim.x + threadIdx.x) >> 5;
  int lane = threadIdx.x & 31;
  if (wid >= Nn) return;
  const float* ar = agg + (size_t)wid * D;
  float z0 = (lane < D)      ? ar[lane]      + b[lane]      : -INFINITY;
  float z1 = (lane + 32 < D) ? ar[lane + 32] + b[lane + 32] : -INFINITY;
  float m = wave_max(fmaxf(z0, z1));
  float p = 0.0f;
  if (lane < D)      p += expf(z0 - m);
  if (lane + 32 < D) p += expf(z1 - m);
  p = wave_sum(p);
  float lg = logf(p);
  float* orow = out + (size_t)wid * D;
  if (lane < D)      orow[lane]      = z0 - m - lg;
  if (lane + 32 < D) orow[lane + 32] = z1 - m - lg;
}

// ---------------------------------------------------------------------------
extern "C" void kernel_launch(void* const* d_in, const int* in_sizes, int n_in,
                              void* d_out, int out_size, void* d_ws, size_t ws_size,
                              hipStream_t stream) {
  const float* x    = (const float*)d_in[0];
  const int*   ei   = (const int*)  d_in[1];
  const float* W0   = (const float*)d_in[2];
  const float* b0   = (const float*)d_in[3];
  const float* W1   = (const float*)d_in[4];
  const float* b1   = (const float*)d_in[5];
  const float* gate = (const float*)d_in[6];
  float* out = (float*)d_out;

  const int HID = in_sizes[3];            // 128
  const int OUT = in_sizes[5];            // 40
  const int INF = in_sizes[2] / HID;      // 128
  const int N   = in_sizes[0] / INF;      // 50000
  const int E   = in_sizes[1] / 2;        // 800000
  const int EN  = E + N;
  const int* rowI = ei;
  const int* colI = ei + E;

  // workspace layout (f32)
  float* w     = (float*)d_ws;
  float* nx    = w; w += N;
  float* nx1   = w; w += N;
  float* ssum  = w; w += N;               // reused as ssum2
  float* degc  = w; w += N;
  float* degw  = w; w += N;
  float* dis   = w; w += N;
  float* sbuf  = w; w += EN;              // cos sims (shared across layers)
  float* ew0   = w; w += EN;              // ew_mem
  float* ewb   = w; w += EN;              // blended layer-1 weights
  float* h     = w; w += (size_t)N * HID; // GEMM output (h0, then h1)
  float* agg   = w; w += (size_t)N * HID; // segment-sum accumulator
  float* x1    = w; w += (size_t)N * HID; // hidden features
  w = (float*)(((uintptr_t)w + 15) & ~(uintptr_t)15);   // 16B align for uint4
  const int tilesM  = (N + 15) / 16;
  const int tilesN0 = (HID + 15) / 16, nk0 = INF >> 5;
  const int tilesN1 = (OUT + 15) / 16, nk1 = HID >> 5;
  unsigned int* Bpack0 = (unsigned int*)w; w += (size_t)tilesN0 * nk0 * 256;
  unsigned int* Bpack1 = (unsigned int*)w; w += (size_t)tilesN1 * nk1 * 256;

  const int B256 = 256;
  auto cdiv = [](long a, long b) { return (int)((a + b - 1) / b); };

  // ---- pre-swizzle both weight matrices for WMMA ----
  pack_B_kernel<<<cdiv((long)tilesN0 * nk0 * 256, B256), B256, 0, stream>>>(W0, Bpack0, INF, HID, tilesN0);
  pack_B_kernel<<<cdiv((long)tilesN1 * nk1 * 256, B256), B256, 0, stream>>>(W1, Bpack1, HID, OUT, tilesN1);

  // ---- layer 0 attention ----
  zero_f32_kernel<<<cdiv(2 * N, B256), B256, 0, stream>>>(ssum, 2 * N); // ssum + degc
  row_norm_kernel<<<cdiv(N, 8), B256, 0, stream>>>(x, nx, N, INF);
  edge_cos_kernel<<<cdiv(E, 8), B256, 0, stream>>>(x, nx, rowI, colI, sbuf, ssum, degc, E, E, INF);
  edge_weight0_kernel<<<cdiv(EN, B256), B256, 0, stream>>>(sbuf, ssum, degc, rowI, ew0, E, N);

  // ---- layer 0 conv ----
  zero_f32_kernel<<<cdiv(N, B256), B256, 0, stream>>>(degw, N);
  deg_accum_kernel<<<cdiv(EN, B256), B256, 0, stream>>>(ew0, colI, degw, EN, E);
  dis_kernel<<<cdiv(N, B256), B256, 0, stream>>>(degw, dis, N);
  {
    int numTiles = tilesM * tilesN0;
    if (nk0 == 4)
      gemm_bf16_wmma_kernel<4><<<cdiv(numTiles, 8), B256, 0, stream>>>(x, Bpack0, h, N, INF, HID,
                                                                       tilesN0, numTiles);
    else
      gemm_bf16_wmma_kernel<0><<<cdiv(numTiles, 8), B256, 0, stream>>>(x, Bpack0, h, N, INF, HID,
                                                                       tilesN0, numTiles);
  }
  zero_f32_kernel<<<cdiv((long)N * HID, B256), B256, 0, stream>>>(agg, N * HID);
  aggregate_kernel<<<cdiv(EN, 8), B256, 0, stream>>>(h, ew0, dis, rowI, colI, agg, EN, E, HID);
  finalize_hidden_kernel<<<cdiv((long)N * HID, B256), B256, 0, stream>>>(agg, b0, x1, N * HID, HID);

  // ---- layer 1 attention (over E+N edges incl. self loops) + gated blend ----
  row_norm_kernel<<<cdiv(N, 8), B256, 0, stream>>>(x1, nx1, N, HID);
  zero_f32_kernel<<<cdiv(N, B256), B256, 0, stream>>>(ssum, N);
  edge_cos_kernel<<<cdiv(EN, 8), B256, 0, stream>>>(x1, nx1, rowI, colI, sbuf, ssum,
                                                    (float*)nullptr, EN, E, HID);
  edge_weight1_kernel<<<cdiv(EN, B256), B256, 0, stream>>>(sbuf, ssum, rowI, ew0, gate, ewb, E, N);

  // ---- layer 1 conv ----
  zero_f32_kernel<<<cdiv(N, B256), B256, 0, stream>>>(degw, N);
  deg_accum_kernel<<<cdiv(EN, B256), B256, 0, stream>>>(ewb, colI, degw, EN, E);
  dis_kernel<<<cdiv(N, B256), B256, 0, stream>>>(degw, dis, N);
  {
    int numTiles = tilesM * tilesN1;
    if (nk1 == 4)
      gemm_bf16_wmma_kernel<4><<<cdiv(numTiles, 8), B256, 0, stream>>>(x1, Bpack1, h, N, HID, OUT,
                                                                       tilesN1, numTiles);
    else
      gemm_bf16_wmma_kernel<0><<<cdiv(numTiles, 8), B256, 0, stream>>>(x1, Bpack1, h, N, HID, OUT,
                                                                       tilesN1, numTiles);
  }
  zero_f32_kernel<<<cdiv((long)N * OUT, B256), B256, 0, stream>>>(agg, N * OUT);
  aggregate_kernel<<<cdiv(EN, 8), B256, 0, stream>>>(h, ewb, dis, rowI, colI, agg, EN, E, OUT);

  // ---- output: + b1, log_softmax ----
  logsoftmax_kernel<<<cdiv(N, 8), B256, 0, stream>>>(agg, b1, out, N, OUT);
}
